// EPA_38448547234132
// MI455X (gfx1250) — matrix operand order repaired and checked by reading, not compiled, not run on hardware
//
#include <hip/hip_runtime.h>
#include <hip/hip_bf16.h>
#include <math.h>

typedef __attribute__((ext_vector_type(16))) __bf16 v16bf;
typedef __attribute__((ext_vector_type(8)))  float  v8f;

#define DEV __device__ __forceinline__

constexpr int Bb = 8, Nn = 8192, Cc = 256, Hh = 8, Dd = 32, Pp = 64;
constexpr int BN = Bb * Nn;   // 65536
constexpr int F4 = 4 * Cc;    // 1024
constexpr int BH = Bb * Hh;   // 64

DEV int lane_id() { return threadIdx.x & 31; }
DEV int wave_id() { return threadIdx.x >> 5; }

DEV v8f wmma_bf16(v16bf a, v16bf b, v8f c) {
  // D = A(16x32 bf16) * B(32x16 bf16) + C(16x16 f32)
  return __builtin_amdgcn_wmma_f32_16x16x32_bf16(false, a, false, b, (short)0, c,
                                                 false, false);
}

// --- CDNA5 async global->LDS DMA (ASYNCcnt-tracked, bypasses VGPRs) ---------
DEV void async_load_b128(const void* lds_dst, const void* gsrc) {
  uint32_t lo = (uint32_t)(uintptr_t)lds_dst;  // LDS aperture: addr[31:0] = DS address
  asm volatile("global_load_async_to_lds_b128 %0, %1, off"
               :: "v"(lo), "v"((uint64_t)(uintptr_t)gsrc)
               : "memory");
}
DEV void async_load_b64(const void* lds_dst, const void* gsrc) {
  uint32_t lo = (uint32_t)(uintptr_t)lds_dst;
  asm volatile("global_load_async_to_lds_b64 %0, %1, off"
               :: "v"(lo), "v"((uint64_t)(uintptr_t)gsrc)
               : "memory");
}
DEV void wait_async0() { asm volatile("s_wait_asynccnt 0" ::: "memory"); }

// ---------------------------------------------------------------- convert ---
__global__ void EPA_cvt_kernel(const float* __restrict__ s, __bf16* __restrict__ d, int n) {
  int i = blockIdx.x * 256 + threadIdx.x;
  int st = gridDim.x * 256;
  for (; i < n; i += st) d[i] = (__bf16)s[i];
}

// --------------------------------------------------------------- qkvv GEMM ---
// qkvvb[BN,1024] = bf16( X[BN,256] @ Wqkvvb[1024,256]^T )
__global__ __launch_bounds__(256)
void EPA_gemm_qkvv_kernel(const float* __restrict__ X, const __bf16* __restrict__ W,
                          __bf16* __restrict__ Q) {
  __shared__ __align__(32) __bf16 As[128][32];
  __shared__ __align__(32) __bf16 Bs[128][32];
  const int m0 = blockIdx.x * 128;
  const int n0 = blockIdx.y * 128;
  const int lane = lane_id(), wid = wave_id();
  const int wr = wid & 1, wc = wid >> 1;              // 2 x 4 waves -> 128x128
  const int b0a = (lane & 16) ? 16 : 0;               // A-frag K segment base
  const int b0b = (lane & 16) ? 16 : 0;               // B-frag K base

  v8f acc[4][2];
#pragma unroll
  for (int i = 0; i < 4; ++i)
#pragma unroll
    for (int j = 0; j < 2; ++j)
#pragma unroll
      for (int e = 0; e < 8; ++e) acc[i][j][e] = 0.f;

  const int lr = threadIdx.x >> 1;        // 0..127
  const int lc = (threadIdx.x & 1) * 16;  // 0 / 16

  for (int k0 = 0; k0 < Cc; k0 += 32) {
    // stage A (fp32 -> bf16 via VALU; conversion must pass through registers)
    const float4* ap = (const float4*)(X + (size_t)(m0 + lr) * Cc + k0 + lc);
#pragma unroll
    for (int j = 0; j < 4; ++j) {
      float4 f = ap[j];
      As[lr][lc + 4 * j + 0] = (__bf16)f.x;
      As[lr][lc + 4 * j + 1] = (__bf16)f.y;
      As[lr][lc + 4 * j + 2] = (__bf16)f.z;
      As[lr][lc + 4 * j + 3] = (__bf16)f.w;
    }
    if (k0 + 32 < Cc)
      __builtin_prefetch(X + (size_t)(m0 + lr) * Cc + k0 + 32 + lc, 0, 1);
    // stage B via async DMA (bf16, no VGPR round-trip)
    const __bf16* bg = W + (size_t)(n0 + lr) * Cc + k0 + lc;
    async_load_b128(&Bs[lr][lc], bg);
    async_load_b128(&Bs[lr][lc + 8], bg + 8);
    wait_async0();
    __syncthreads();
#pragma unroll
    for (int fm = 0; fm < 4; ++fm) {
      const __bf16* arow = &As[wr * 64 + fm * 16 + (lane & 15)][0];
      v16bf a;
#pragma unroll
      for (int e = 0; e < 8; ++e) { a[e] = arow[b0a + e]; a[e + 8] = arow[b0a + 8 + e]; }
#pragma unroll
      for (int fn = 0; fn < 2; ++fn) {
        const __bf16* brow = &Bs[wc * 32 + fn * 16 + (lane & 15)][0];
        v16bf b = *(const v16bf*)&brow[b0b];
        acc[fm][fn] = wmma_bf16(a, b, acc[fm][fn]);
      }
    }
    __syncthreads();
  }
  const int hi8 = (lane & 16) ? 8 : 0;
#pragma unroll
  for (int fm = 0; fm < 4; ++fm)
#pragma unroll
    for (int fn = 0; fn < 2; ++fn)
#pragma unroll
      for (int r = 0; r < 8; ++r) {
        int row = m0 + wr * 64 + fm * 16 + r + hi8;
        int col = n0 + wc * 32 + fn * 16 + (lane & 15);
        Q[(size_t)row * F4 + col] = (__bf16)acc[fm][fn][r];
      }
}

// ------------------------------------------------------------------- norms ---
// inv_n[b,h*32+d] = 1/max(||q[b,h,d,:]||, eps) ; same for k. 4096 blocks.
__global__ __launch_bounds__(256)
void EPA_norms_kernel(const __bf16* __restrict__ Q, float* __restrict__ inv_nq,
                      float* __restrict__ inv_nk) {
  int idx = blockIdx.x;
  int part = idx >> 11;     // 0: q, 1: k
  int j = idx & 2047;       // b*256 + h*32 + d
  int b = j >> 8;
  int hd = j & 255;
  const __bf16* base = Q + (size_t)b * Nn * F4 + part * Cc + hd;
  float s = 0.f;
  for (int n = threadIdx.x; n < Nn; n += 256) {
    float v = (float)base[(size_t)n * F4];
    s += v * v;
  }
  __shared__ float red[256];
  red[threadIdx.x] = s;
  __syncthreads();
  for (int o = 128; o > 0; o >>= 1) {
    if (threadIdx.x < o) red[threadIdx.x] += red[threadIdx.x + o];
    __syncthreads();
  }
  if (threadIdx.x == 0) {
    float nrm = fmaxf(sqrtf(red[0]), 1e-12f);
    (part ? inv_nk : inv_nq)[j] = 1.0f / nrm;
  }
}

// -------------------------------------------------------- E / F projections ---
// part 0: kprojT[bh][p][d] = bf16( sum_n k[d,n]*WE[p,n] + bE[p] )
// part 1: vproj [bh][d][p] = bf16( sum_n vsa[d,n]*WF[p,n] + bF[p] )
__global__ __launch_bounds__(256)
void EPA_proj_kernel(const __bf16* __restrict__ Q, const __bf16* __restrict__ WEb,
                     const __bf16* __restrict__ WFb, const float* __restrict__ bE,
                     const float* __restrict__ bF, __bf16* __restrict__ kprojT,
                     __bf16* __restrict__ vproj) {
  const int bh = blockIdx.x;
  const int part = blockIdx.y;
  const int b = bh >> 3, h = bh & 7;
  const int colbase = (part ? 768 : 256) + h * 32;
  const __bf16* Wm = part ? WFb : WEb;
  const float* bias = part ? bF : bE;

  __shared__ __align__(32) __bf16 T[32][32];  // [n][d] token tile
  const int lane = lane_id(), wid = wave_id();
  const int fm = wid & 1, fn = wid >> 1;  // 2 x 4 waves -> 32x64
  const int b0a = (lane & 16) ? 16 : 0;
  const int b0b = (lane & 16) ? 16 : 0;

  v8f acc;
#pragma unroll
  for (int e = 0; e < 8; ++e) acc[e] = 0.f;

  const int ln = threadIdx.x >> 3;        // 0..31 token row
  const int ld = (threadIdx.x & 7) * 4;   // 0..28 dim col

  for (int k0 = 0; k0 < Nn; k0 += 32) {
    const __bf16* src = Q + (size_t)(b * Nn + k0 + ln) * F4 + colbase + ld;
    async_load_b64(&T[ln][ld], src);      // 4 halves = 8B per lane via DMA
    wait_async0();
    __syncthreads();
    int drow = fm * 16 + (lane & 15);
    v16bf a;  // A[d][n] (transposed read from LDS)
#pragma unroll
    for (int e = 0; e < 8; ++e) { a[e] = T[b0a + e][drow]; a[e + 8] = T[b0a + 8 + e][drow]; }
    int p = fn * 16 + (lane & 15);
    v16bf bfr = *(const v16bf*)(Wm + (size_t)p * Nn + k0 + b0b);
    acc = wmma_bf16(a, bfr, acc);
    __syncthreads();
  }
  const int hi8 = (lane & 16) ? 8 : 0;
#pragma unroll
  for (int r = 0; r < 8; ++r) {
    int d = fm * 16 + r + hi8;
    int p = fn * 16 + (lane & 15);
    float v = acc[r] + bias[p];
    if (part == 0) kprojT[(size_t)bh * 2048 + p * 32 + d] = (__bf16)v;
    else           vproj [(size_t)bh * 2048 + d * 64 + p] = (__bf16)v;
  }
}

// ------------------------------------------------------- channel attention ---
__global__ __launch_bounds__(128)
void EPA_chan_attn_kernel(const __bf16* __restrict__ Q, const float* __restrict__ inv_nq,
                          const float* __restrict__ inv_nk, const float* __restrict__ temp1,
                          __bf16* __restrict__ xca) {
  const int bh = blockIdx.x;
  const int b = bh >> 3, h = bh & 7;
  __shared__ __align__(32) __bf16 qt[32][32];
  __shared__ __align__(32) __bf16 kt[32][32];
  __shared__ float attnS[32][32];
  __shared__ __align__(32) __bf16 attnP[32][32];
  const int lane = lane_id(), wid = wave_id();
  const int fm = wid & 1, fn = wid >> 1;  // 2x2 waves -> 32x32
  const int b0a = (lane & 16) ? 16 : 0, b0b = (lane & 16) ? 16 : 0;
  v8f acc;
#pragma unroll
  for (int e = 0; e < 8; ++e) acc[e] = 0.f;

  const int ln = threadIdx.x >> 2;        // 0..31
  const int ld = (threadIdx.x & 3) * 8;   // 0,8,16,24
  const int qcol = h * 32, kcol = 256 + h * 32;
  for (int k0 = 0; k0 < Nn; k0 += 32) {
    const __bf16* qsrc = Q + (size_t)(b * Nn + k0 + ln) * F4 + qcol + ld;
    const __bf16* ksrc = Q + (size_t)(b * Nn + k0 + ln) * F4 + kcol + ld;
    async_load_b128(&qt[ln][ld], qsrc);   // 8 halves = 16B
    async_load_b128(&kt[ln][ld], ksrc);
    wait_async0();
    __syncthreads();
    int drow = fm * 16 + (lane & 15);
    v16bf a;
#pragma unroll
    for (int e = 0; e < 8; ++e) { a[e] = qt[b0a + e][drow]; a[e + 8] = qt[b0a + 8 + e][drow]; }
    int ecol = fn * 16 + (lane & 15);
    v16bf bb;
#pragma unroll
    for (int e = 0; e < 16; ++e) bb[e] = kt[b0b + e][ecol];
    acc = wmma_bf16(a, bb, acc);
    __syncthreads();
  }
  const int hi8 = (lane & 16) ? 8 : 0;
  float t1 = temp1[h];
#pragma unroll
  for (int r = 0; r < 8; ++r) {
    int d = fm * 16 + r + hi8, e = fn * 16 + (lane & 15);
    attnS[d][e] = acc[r] * inv_nq[b * 256 + h * 32 + d] * inv_nk[b * 256 + h * 32 + e] * t1;
  }
  __syncthreads();
  if (threadIdx.x < 32) {
    int d = threadIdx.x;
    float mx = -1e30f;
    for (int e = 0; e < 32; ++e) mx = fmaxf(mx, attnS[d][e]);
    float s = 0.f;
    for (int e = 0; e < 32; ++e) { float ex = __expf(attnS[d][e] - mx); attnS[d][e] = ex; s += ex; }
    float inv = 1.f / s;
    for (int e = 0; e < 32; ++e) attnP[d][e] = (__bf16)(attnS[d][e] * inv);
  }
  __syncthreads();
  // x_ca[d,n] = attnP @ v_ca ; B fragments read K-contiguous from natural layout
  for (int n0 = 0; n0 < Nn; n0 += 64) {
    int ncol = n0 + wid * 16 + (lane & 15);
    const __bf16* vsrc = Q + (size_t)(b * Nn + ncol) * F4 + 512 + h * 32 + b0b;
    v16bf bb = *(const v16bf*)vsrc;
#pragma unroll
    for (int fm2 = 0; fm2 < 2; ++fm2) {
      const __bf16* arow = &attnP[fm2 * 16 + (lane & 15)][0];
      v16bf a;
#pragma unroll
      for (int e = 0; e < 8; ++e) { a[e] = arow[b0a + e]; a[e + 8] = arow[b0a + 8 + e]; }
      v8f o;
#pragma unroll
      for (int e = 0; e < 8; ++e) o[e] = 0.f;
      o = wmma_bf16(a, bb, o);
#pragma unroll
      for (int r = 0; r < 8; ++r) {
        int d = fm2 * 16 + r + hi8;
        xca[(size_t)(b * Nn + ncol) * Cc + h * 32 + d] = (__bf16)o[r];
      }
    }
  }
}

// ------------------------------------------------------- spatial attention ---
__global__ __launch_bounds__(128)
void EPA_spat_attn_kernel(const __bf16* __restrict__ Q, const __bf16* __restrict__ kprojT,
                          const __bf16* __restrict__ vproj, const float* __restrict__ inv_nq,
                          const float* __restrict__ temp2, __bf16* __restrict__ xsa) {
  const int bh = blockIdx.x;
  const int b = bh >> 3, h = bh & 7;
  const int tok0 = blockIdx.y * 64;
  __shared__ float invq[32];
  __shared__ float attnS[4][16][64];
  __shared__ __align__(32) __bf16 attnP[4][16][64];
  if (threadIdx.x < 32) invq[threadIdx.x] = inv_nq[b * 256 + h * 32 + threadIdx.x];
  __syncthreads();
  const int lane = lane_id(), wid = wave_id();
  const int b0a = (lane & 16) ? 16 : 0, b0b = (lane & 16) ? 16 : 0;
  const int myTok = tok0 + wid * 16;
  const int hi8 = (lane & 16) ? 8 : 0;

  // phase 1: attn[n,p] = (q/||q||)^T @ k_proj  (K = d = 32, one WMMA per p-frag)
  const __bf16* qrow = Q + (size_t)(b * Nn + myTok + (lane & 15)) * F4 + h * 32;
  v16bf a;
#pragma unroll
  for (int e = 0; e < 8; ++e) {
    a[e]     = (__bf16)((float)qrow[b0a + e]      * invq[b0a + e]);
    a[e + 8] = (__bf16)((float)qrow[b0a + 8 + e] * invq[b0a + 8 + e]);
  }
  float t2 = temp2[h];
#pragma unroll
  for (int fn = 0; fn < 4; ++fn) {
    int p = fn * 16 + (lane & 15);
    v16bf bb = *(const v16bf*)(kprojT + (size_t)bh * 2048 + p * 32 + b0b);
    v8f o;
#pragma unroll
    for (int e = 0; e < 8; ++e) o[e] = 0.f;
    o = wmma_bf16(a, bb, o);
#pragma unroll
    for (int r = 0; r < 8; ++r) attnS[wid][r + hi8][p] = o[r] * t2;
  }
  __syncthreads();
  if (lane < 16) {
    int m = lane;
    float mx = -1e30f;
    for (int p = 0; p < Pp; ++p) mx = fmaxf(mx, attnS[wid][m][p]);
    float s = 0.f;
    for (int p = 0; p < Pp; ++p) { float ex = __expf(attnS[wid][m][p] - mx); attnS[wid][m][p] = ex; s += ex; }
    float inv = 1.f / s;
    for (int p = 0; p < Pp; ++p) attnP[wid][m][p] = (__bf16)(attnS[wid][m][p] * inv);
  }
  __syncthreads();
  // phase 2: x_sa[n,d] = attnP @ v_proj^T (K = p = 64)
#pragma unroll
  for (int fd = 0; fd < 2; ++fd) {
    v8f o;
#pragma unroll
    for (int e = 0; e < 8; ++e) o[e] = 0.f;
#pragma unroll
    for (int c = 0; c < 2; ++c) {
      const __bf16* arow = &attnP[wid][lane & 15][c * 32];
      v16bf a2;
#pragma unroll
      for (int e = 0; e < 8; ++e) { a2[e] = arow[b0a + e]; a2[e + 8] = arow[b0a + 8 + e]; }
      int d = fd * 16 + (lane & 15);
      v16bf bb = *(const v16bf*)(vproj + (size_t)bh * 2048 + d * 64 + c * 32 + b0b);
      o = wmma_bf16(a2, bb, o);
    }
    // faithful scrambled store: row = d*256 + h*32 + (n>>8), col = n & 255
#pragma unroll
    for (int r = 0; r < 8; ++r) {
      int n = myTok + r + hi8;
      int d = fd * 16 + (lane & 15);
      int rr = d * 256 + h * 32 + (n >> 8);
      int cc = n & 255;
      xsa[(size_t)b * Nn * Cc + (size_t)rr * Cc + cc] = (__bf16)o[r];
    }
  }
}

// ------------------------------------------------------------- output GEMMs ---
__global__ __launch_bounds__(256)
void EPA_gemm_out_kernel(const __bf16* __restrict__ Xsa, const __bf16* __restrict__ Xca,
                         const __bf16* __restrict__ Wo1b, const __bf16* __restrict__ Wo2b,
                         const float* __restrict__ bo1, const float* __restrict__ bo2,
                         float* __restrict__ out) {
  const int m0 = blockIdx.x * 128;
  const int branch = blockIdx.y;
  const __bf16* A = branch ? Xca : Xsa;
  const __bf16* W = branch ? Wo2b : Wo1b;
  const float* bias = branch ? bo2 : bo1;
  const int coloff = branch ? 128 : 0;
  __shared__ __align__(32) __bf16 As[128][32];
  __shared__ __align__(32) __bf16 Bs[128][32];
  const int lane = lane_id(), wid = wave_id();
  const int wr = wid & 1, wc = wid >> 1;
  const int b0a = (lane & 16) ? 16 : 0, b0b = (lane & 16) ? 16 : 0;
  v8f acc[4][2];
#pragma unroll
  for (int i = 0; i < 4; ++i)
#pragma unroll
    for (int j = 0; j < 2; ++j)
#pragma unroll
      for (int e = 0; e < 8; ++e) acc[i][j][e] = 0.f;
  const int lr = threadIdx.x >> 1;
  const int lc = (threadIdx.x & 1) * 16;
  for (int k0 = 0; k0 < Cc; k0 += 32) {
    const __bf16* ag = A + (size_t)(m0 + lr) * Cc + k0 + lc;
    const __bf16* bg = W + (size_t)lr * Cc + k0 + lc;
    async_load_b128(&As[lr][lc], ag);
    async_load_b128(&As[lr][lc + 8], ag + 8);
    async_load_b128(&Bs[lr][lc], bg);
    async_load_b128(&Bs[lr][lc + 8], bg + 8);
    wait_async0();
    __syncthreads();
#pragma unroll
    for (int fm = 0; fm < 4; ++fm) {
      const __bf16* arow = &As[wr * 64 + fm * 16 + (lane & 15)][0];
      v16bf a;
#pragma unroll
      for (int e = 0; e < 8; ++e) { a[e] = arow[b0a + e]; a[e + 8] = arow[b0a + 8 + e]; }
#pragma unroll
      for (int fn = 0; fn < 2; ++fn) {
        const __bf16* brow = &Bs[wc * 32 + fn * 16 + (lane & 15)][0];
        v16bf bb = *(const v16bf*)&brow[b0b];
        acc[fm][fn] = wmma_bf16(a, bb, acc[fm][fn]);
      }
    }
    __syncthreads();
  }
  const int hi8 = (lane & 16) ? 8 : 0;
#pragma unroll
  for (int fm = 0; fm < 4; ++fm)
#pragma unroll
    for (int fn = 0; fn < 2; ++fn)
#pragma unroll
      for (int r = 0; r < 8; ++r) {
        int row = m0 + wr * 64 + fm * 16 + r + hi8;
        int col = wc * 32 + fn * 16 + (lane & 15);
        out[(size_t)row * Cc + coloff + col] = acc[fm][fn][r] + bias[col];
      }
}

// ------------------------------------------------------------------ driver ---
extern "C" void kernel_launch(void* const* d_in, const int* in_sizes, int n_in,
                              void* d_out, int out_size, void* d_ws, size_t ws_size,
                              hipStream_t stream) {
  const float* x     = (const float*)d_in[0];
  const float* Wqkvv = (const float*)d_in[1];
  const float* WE    = (const float*)d_in[2];
  const float* bE    = (const float*)d_in[3];
  const float* WF    = (const float*)d_in[4];
  const float* bF    = (const float*)d_in[5];
  const float* Wo1   = (const float*)d_in[6];
  const float* bo1   = (const float*)d_in[7];
  const float* Wo2   = (const float*)d_in[8];
  const float* bo2   = (const float*)d_in[9];
  const float* temp1 = (const float*)d_in[10];
  const float* temp2 = (const float*)d_in[11];
  float* out = (float*)d_out;

  char* ws = (char*)d_ws;
  size_t off = 0;
  auto alloc = [&](size_t bytes) -> char* {
    char* p = ws + off;
    off += (bytes + 255) & ~(size_t)255;
    return p;
  };
  __bf16* qkvvb  = (__bf16*)alloc((size_t)BN * F4 * 2);      // 128 MB
  __bf16* Wqkvvb = (__bf16*)alloc((size_t)F4 * Cc * 2);
  __bf16* WEb    = (__bf16*)alloc((size_t)Pp * Nn * 2);
  __bf16* WFb    = (__bf16*)alloc((size_t)Pp * Nn * 2);
  __bf16* Wo1b   = (__bf16*)alloc((size_t)128 * Cc * 2);
  __bf16* Wo2b   = (__bf16*)alloc((size_t)128 * Cc * 2);
  __bf16* kprojT = (__bf16*)alloc((size_t)BH * Pp * Dd * 2);
  __bf16* vproj  = (__bf16*)alloc((size_t)BH * Dd * Pp * 2);
  __bf16* xsa    = (__bf16*)alloc((size_t)BN * Cc * 2);      // 32 MB
  __bf16* xca    = (__bf16*)alloc((size_t)BN * Cc * 2);      // 32 MB
  float*  inv_nq = (float*)alloc((size_t)Bb * Cc * 4);
  float*  inv_nk = (float*)alloc((size_t)Bb * Cc * 4);

  EPA_cvt_kernel<<<256, 256, 0, stream>>>(Wqkvv, Wqkvvb, F4 * Cc);
  EPA_cvt_kernel<<<512, 256, 0, stream>>>(WE, WEb, Pp * Nn);
  EPA_cvt_kernel<<<512, 256, 0, stream>>>(WF, WFb, Pp * Nn);
  EPA_cvt_kernel<<<32, 256, 0, stream>>>(Wo1, Wo1b, 128 * Cc);
  EPA_cvt_kernel<<<32, 256, 0, stream>>>(Wo2, Wo2b, 128 * Cc);

  EPA_gemm_qkvv_kernel<<<dim3(BN / 128, F4 / 128), 256, 0, stream>>>(x, Wqkvvb, qkvvb);
  EPA_norms_kernel<<<2 * Bb * Cc, 256, 0, stream>>>(qkvvb, inv_nq, inv_nk);
  EPA_proj_kernel<<<dim3(BH, 2), 256, 0, stream>>>(qkvvb, WEb, WFb, bE, bF, kprojT, vproj);
  EPA_chan_attn_kernel<<<BH, 128, 0, stream>>>(qkvvb, inv_nq, inv_nk, temp1, xca);
  EPA_spat_attn_kernel<<<dim3(BH, Nn / 64), 128, 0, stream>>>(qkvvb, kprojT, vproj, inv_nq,
                                                              temp2, xsa);
  EPA_gemm_out_kernel<<<dim3(BN / 128, 2), 256, 0, stream>>>(xsa, xca, Wo1b, Wo2b, bo1, bo2,
                                                             out);
}
